// EnhancedGenerativeFusionReranker_24343874634404
// MI455X (gfx1250) — matrix operand order repaired and checked by reading, not compiled, not run on hardware
//
#include <hip/hip_runtime.h>
#include <math.h>

typedef __attribute__((ext_vector_type(16))) _Float16 v16h;
typedef __attribute__((ext_vector_type(8)))  float    v8f;
typedef __attribute__((ext_vector_type(4)))  int      v4i;

#define NCAND 131072
#define DIM   512
#define NM    3
#define HDIM  256
#define H2DIM 128
#define CDIM  64
#define NGEN  20
#define ROWS  64      // candidate rows per block (16 per wave * 4 waves)
#define TPB   128
#define ASTR  520     // halves, padded row stride of candidate tile
#define BSTR  48      // halves, padded row stride of B^T panel
#define H1STR 264     // halves
#define FSTR  260     // floats
#define SASTR 48      // halves

// ---- LDS region map (bytes) ----
#define OFF_A    0        // 64*520*2 = 66560   (f16 A tile; later f32 staging FSTR)
#define OFF_B0   66560    // 256*48*2 = 24576   (B^T K-panel, buffer 0)
#define OFF_B1   91136    // 256*48*2 = 24576   (B^T K-panel, buffer 1)
#define BBYTES   24576
#define OFF_H1   115712   // 64*264*2 = 33792   (f16 hidden; overlapped by sims stage)
#define OFF_S    115712   // 64*24*4  = 6144    (sims staging, dead before H1 written)
#define OFF_NORM 149504   // 64 f32
#define OFF_PART 149760   // 128 f32
#define OFF_GEN  150272   // 64 f32
#define OFF_PERS 150528   // 64 f32
#define OFF_NOV  150784   // 64 f32
#define OFF_FIN  151040   // 64 f32
#define OFF_CONF 151296   // 64 f32
#define OFF_SA   151552   // 64*48*2 = 6144    (padded score matrix, K=32)
#define SMEM_BYTES 157696

struct KParams {
  const float *q, *cand, *user, *ctx, *trad, *gen;
  const float *ml_w1, *ml_b1, *ml_w2, *ml_b2;
  const float *nv_w1, *nv_b1, *nv_w2, *nv_b2, *nv_w3, *nv_b3;
  const float *sf_w1, *sf_b1, *sf_g1, *sf_be1;
  const float *sf_w2, *sf_b2, *sf_g2, *sf_be2, *sf_w3, *sf_b3;
  const float *cc_w1, *cc_b1, *cc_w2, *cc_b2;
  _Float16 *w1cT, *w2T, *gT, *sf1T, *sf2T, *cc1T;   // f16 transposed weights in ws
  float *qh1, *invg, *tsum;                          // ws scalars
  float *out;
};

// ---- WMMA helper ----
static __device__ inline v8f wmma_f16(v16h a, v16h b, v8f c) {
  return __builtin_amdgcn_wmma_f32_16x16x32_f16(false, a, false, b, (short)0, c, false, false);
}

// ---- CDNA5 async global->LDS copy (16B per lane), ASYNCcnt-tracked ----
#if defined(__gfx1250__) && __has_builtin(__builtin_amdgcn_global_load_async_to_lds_b128)
#define HAVE_ASYNC_LDS 1
#else
#define HAVE_ASYNC_LDS 0
#endif

static __device__ inline void async_copy16(const _Float16* g, _Float16* l) {
#if HAVE_ASYNC_LDS
  __builtin_amdgcn_global_load_async_to_lds_b128(
      (__attribute__((address_space(1))) v4i*)g,
      (__attribute__((address_space(3))) v4i*)l, 0, 0);
#else
  *(uint4*)l = *(const uint4*)g;
#endif
}

static __device__ inline void async_wait() {
#if HAVE_ASYNC_LDS
#if __has_builtin(__builtin_amdgcn_s_wait_asynccnt)
  __builtin_amdgcn_s_wait_asynccnt(0);
#else
  asm volatile("s_wait_asynccnt 0" ::: "memory");
#endif
#endif
}

// A fragment (16x32 f16): lanes 0-15 -> M=lane, K=kb+[0..7],[16..23];
// lanes 16-31 -> M=lane-16, K=kb+[8..15],[24..31]
static __device__ inline v16h frag_a(const _Float16* m, int strideH, int mrow, int kb, int lane) {
  int r  = mrow + (lane & 15);
  int c0 = kb + ((lane < 16) ? 0 : 8);
  const char* p = (const char*)(m + r * strideH + c0);
  union { v16h v; uint4 q[2]; } f;
  f.q[0] = *(const uint4*)p;
  f.q[1] = *(const uint4*)(p + 32);
  return f.v;
}

// B fragment (32x16 f16) from B^T (row n = output column, contiguous K):
// lanes 0-15 -> N=lane, K=kb+[0..7],[16..23]; lanes 16-31 -> N=lane-16, K=kb+[8..15],[24..31]
static __device__ inline v16h frag_bt(const _Float16* bT, int strideH, int kb, int n0, int lane) {
  int n  = n0 + (lane & 15);
  int kk = kb + ((lane < 16) ? 0 : 8);
  const char* p = (const char*)(bT + n * strideH + kk);
  union { v16h v; uint4 q[2]; } f;
  f.q[0] = *(const uint4*)p;
  f.q[1] = *(const uint4*)(p + 32);
  return f.v;
}

// ============================ prep kernels ============================
__global__ void k_zero(KParams kp) {
  if (threadIdx.x < 4) kp.tsum[threadIdx.x] = 0.f;
}

__global__ __launch_bounds__(256) void k_tsum(KParams kp) {
  __shared__ float red[256];
  float p0 = 0.f, p1 = 0.f, p2 = 0.f;
  for (size_t i = (size_t)blockIdx.x * blockDim.x + threadIdx.x; i < (size_t)NCAND;
       i += (size_t)gridDim.x * blockDim.x) {
    p0 += kp.trad[i * 3 + 0]; p1 += kp.trad[i * 3 + 1]; p2 += kp.trad[i * 3 + 2];
  }
  float ps[3] = {p0, p1, p2};
  for (int m = 0; m < 3; ++m) {
    red[threadIdx.x] = ps[m]; __syncthreads();
    for (int s = 128; s > 0; s >>= 1) {
      if ((int)threadIdx.x < s) red[threadIdx.x] += red[threadIdx.x + s];
      __syncthreads();
    }
    if (threadIdx.x == 0) atomicAdd(&kp.tsum[m], red[0]);
    __syncthreads();
  }
}

__global__ __launch_bounds__(256) void k_norms(KParams kp) {
  __shared__ float red[256];
  int b = blockIdx.x;
  const float* v = (b < NGEN) ? (kp.gen + (size_t)b * DIM) : kp.user;
  float s = 0.f;
  for (int c = threadIdx.x; c < DIM; c += 256) { float x = v[c]; s += x * x; }
  red[threadIdx.x] = s; __syncthreads();
  for (int st = 128; st > 0; st >>= 1) {
    if ((int)threadIdx.x < st) red[threadIdx.x] += red[threadIdx.x + st];
    __syncthreads();
  }
  if (threadIdx.x == 0) kp.invg[b] = 1.f / fmaxf(sqrtf(red[0]), 1e-8f);
}

__global__ __launch_bounds__(256) void k_prep(KParams kp) {
  __shared__ float h[HDIM];
  __shared__ float lg[3];
  int n = threadIdx.x;
  // qh1[n] = nv_b1[n] + q . nv_w1[0:512, n]   (query half of novelty L1, constant per row)
  {
    float s = kp.nv_b1[n];
    for (int k = 0; k < DIM; ++k) s += kp.q[k] * kp.nv_w1[(size_t)k * HDIM + n];
    kp.qh1[n] = s;
  }
  // meta-learner hidden
  {
    float s = kp.ml_b1[n];
    for (int k = 0; k < DIM; ++k) s += kp.user[k] * kp.ml_w1[(size_t)k * HDIM + n];
    for (int m = 0; m < 3; ++m)
      s += (kp.tsum[m] * (1.f / (float)NCAND)) * kp.ml_w1[(size_t)(DIM + m) * HDIM + n];
    for (int c = 0; c < CDIM; ++c)
      s += kp.ctx[c] * kp.ml_w1[(size_t)(DIM + 3 + c) * HDIM + n];
    h[n] = fmaxf(s, 0.f);
  }
  __syncthreads();
  if (n < 3) {
    float s = kp.ml_b2[n];
    for (int k = 0; k < HDIM; ++k) s += h[k] * kp.ml_w2[k * 3 + n];
    lg[n] = s;
  }
  __syncthreads();
  if (n == 0) {
    float mx = fmaxf(lg[0], fmaxf(lg[1], lg[2]));
    float e0 = expf(lg[0] - mx), e1 = expf(lg[1] - mx), e2 = expf(lg[2] - mx);
    float inv = 1.f / (e0 + e1 + e2);
    kp.out[5 * (size_t)NCAND + 0] = e0 * inv;
    kp.out[5 * (size_t)NCAND + 1] = e1 * inv;
    kp.out[5 * (size_t)NCAND + 2] = e2 * inv;
  }
}

// transpose + f16-convert all GEMM weights into workspace
__global__ __launch_bounds__(256) void k_convert(KParams kp) {
  for (size_t i = (size_t)blockIdx.x * 256 + threadIdx.x; i < 225280;
       i += (size_t)gridDim.x * 256) {
    if (i < 131072) {                       // w1cT[256][512] <- nv_w1[512+k][n]
      size_t n = i >> 9, k = i & 511;
      kp.w1cT[n * DIM + k] = (_Float16)kp.nv_w1[(DIM + k) * HDIM + n];
    } else if (i < 163840) {                // w2T[128][256]
      size_t j = i - 131072, n = j >> 8, k = j & 255;
      kp.w2T[n * HDIM + k] = (_Float16)kp.nv_w2[k * H2DIM + n];
    } else if (i < 180224) {                // gT[32][512]: 20 unit gens, unit user, zero pad
      size_t j = i - 163840, n = j >> 9, k = j & 511;
      float v = 0.f;
      if (n < NGEN)       v = kp.gen[n * DIM + k] * kp.invg[n];
      else if (n == NGEN) v = kp.user[k] * kp.invg[NGEN];
      kp.gT[n * DIM + k] = (_Float16)v;
    } else if (i < 188416) {                // sf1T[256][32], K padded 5->32
      size_t j = i - 180224, n = j >> 5, k = j & 31;
      kp.sf1T[n * 32 + k] = (_Float16)((k < 5) ? kp.sf_w1[k * HDIM + n] : 0.f);
    } else if (i < 221184) {                // sf2T[128][256]
      size_t j = i - 188416, n = j >> 8, k = j & 255;
      kp.sf2T[n * HDIM + k] = (_Float16)kp.sf_w2[k * H2DIM + n];
    } else {                                // cc1T[128][32], K padded 5->32
      size_t j = i - 221184, n = j >> 5, k = j & 31;
      kp.cc1T[n * 32 + k] = (_Float16)((k < 5) ? kp.cc_w1[k * H2DIM + n] : 0.f);
    }
  }
}

// ============================ fused main kernel ============================
__global__ __launch_bounds__(TPB, 1) void k_main(KParams kp) {
  __shared__ alignas(16) char smem[SMEM_BYTES];
  const int tid  = threadIdx.x;
  const int lane = tid & 31;
  const int w    = tid >> 5;            // wave id 0..3; wave handles rows 16w..16w+15
  const int r0   = blockIdx.x * ROWS;

  _Float16* ldsA  = (_Float16*)(smem + OFF_A);
  _Float16* ldsH1 = (_Float16*)(smem + OFF_H1);
  _Float16* ldsSA = (_Float16*)(smem + OFF_SA);
  float* normf = (float*)(smem + OFF_NORM);
  float* partf = (float*)(smem + OFF_PART);
  float* genf  = (float*)(smem + OFF_GEN);
  float* persf = (float*)(smem + OFF_PERS);
  float* novf  = (float*)(smem + OFF_NOV);
  float* finf  = (float*)(smem + OFF_FIN);
  float* conff = (float*)(smem + OFF_CONF);

  // ---- P0: load candidate tile, convert to f16, row norms ----
  for (int i = tid; i < ROWS * DIM; i += TPB) {
    int row = i >> 9, col = i & 511;
    ldsA[row * ASTR + col] = (_Float16)kp.cand[(size_t)(r0 + row) * DIM + col];
  }
  __syncthreads();
  {
    int row = tid >> 1, hh = tid & 1;
    const _Float16* p = ldsA + row * ASTR + hh * 256;
    float s = 0.f;
    for (int c = 0; c < 256; ++c) { float x = (float)p[c]; s += x * x; }
    partf[tid] = s;
  }
  __syncthreads();
  if (tid < ROWS) normf[tid] = sqrtf(partf[2 * tid] + partf[2 * tid + 1]);
  __syncthreads();

  // ---- P1: gens GEMM [64x512]x[512x32] -> max-sim + personalization ----
  {
    v8f a0 = (v8f)0.0f, a1 = (v8f)0.0f;
    for (int ks = 0; ks < 16; ++ks) {
      v16h a  = frag_a(ldsA, ASTR, 16 * w, ks * 32, lane);
      v16h b0 = frag_bt(kp.gT, DIM, ks * 32, 0, lane);
      v16h b1 = frag_bt(kp.gT, DIM, ks * 32, 16, lane);
      a0 = wmma_f16(a, b0, a0);
      a1 = wmma_f16(a, b1, a1);
    }
    float* ldsS = (float*)(smem + OFF_S);
#pragma unroll
    for (int j = 0; j < 8; ++j) {
      int row = 16 * w + j + ((lane < 16) ? 0 : 8);
      int n0  = lane & 15;
      ldsS[row * 24 + n0] = a0[j];
      if (16 + n0 < 21) ldsS[row * 24 + 16 + n0] = a1[j];
    }
  }
  __syncthreads();
  if (tid < ROWS) {
    float denom = fmaxf(normf[tid], 1e-8f);
    const float* s = (const float*)(smem + OFF_S) + tid * 24;
    float g = s[0];
    for (int n = 1; n < NGEN; ++n) g = fmaxf(g, s[n]);
    genf[tid]  = g / denom;
    persf[tid] = s[20] / denom;
  }
  __syncthreads();

  // ---- P2: novelty L1 [64x512]x[512x256], async double-buffered B panels ----
  {
    v8f acc[16];
#pragma unroll
    for (int t = 0; t < 16; ++t) acc[t] = (v8f)0.0f;
    // preload panel 0
    for (int i = tid; i < 1024; i += TPB) {
      int n = i >> 2, seg = i & 3;
      async_copy16(kp.w1cT + n * DIM + seg * 8,
                   (_Float16*)(smem + OFF_B0 + n * 96 + seg * 16));
    }
    async_wait();
    __syncthreads();
    for (int ks = 0; ks < 16; ++ks) {
      _Float16* cur = (_Float16*)(smem + OFF_B0 + (ks & 1) * BBYTES);
      if (ks < 15) {
        char* nxt = smem + OFF_B0 + ((ks + 1) & 1) * BBYTES;
        for (int i = tid; i < 1024; i += TPB) {
          int n = i >> 2, seg = i & 3;
          async_copy16(kp.w1cT + n * DIM + (ks + 1) * 32 + seg * 8,
                       (_Float16*)(nxt + n * 96 + seg * 16));
        }
      }
      v16h a = frag_a(ldsA, ASTR, 16 * w, ks * 32, lane);
#pragma unroll
      for (int t = 0; t < 16; ++t) {
        v16h b = frag_bt(cur, BSTR, 0, 16 * t, lane);
        acc[t] = wmma_f16(a, b, acc[t]);
      }
      if (ks < 15) { async_wait(); __syncthreads(); }
    }
    __syncthreads();
#pragma unroll
    for (int t = 0; t < 16; ++t) {
      int col  = 16 * t + (lane & 15);
      float qb = kp.qh1[col];
#pragma unroll
      for (int j = 0; j < 8; ++j) {
        int row = 16 * w + j + ((lane < 16) ? 0 : 8);
        ldsH1[row * H1STR + col] = (_Float16)fmaxf(acc[t][j] + qb, 0.f);
      }
    }
  }
  __syncthreads();

  // ---- P3: novelty L2 [64x256]x[256x128] + sigmoid head ----
  {
    v8f acc[8];
#pragma unroll
    for (int t = 0; t < 8; ++t) acc[t] = (v8f)0.0f;
    for (int i = tid; i < 512; i += TPB) {
      int n = i >> 2, seg = i & 3;
      async_copy16(kp.w2T + n * HDIM + seg * 8,
                   (_Float16*)(smem + OFF_B0 + n * 96 + seg * 16));
    }
    async_wait();
    __syncthreads();
    for (int ks = 0; ks < 8; ++ks) {
      _Float16* cur = (_Float16*)(smem + OFF_B0 + (ks & 1) * BBYTES);
      if (ks < 7) {
        char* nxt = smem + OFF_B0 + ((ks + 1) & 1) * BBYTES;
        for (int i = tid; i < 512; i += TPB) {
          int n = i >> 2, seg = i & 3;
          async_copy16(kp.w2T + n * HDIM + (ks + 1) * 32 + seg * 8,
                       (_Float16*)(nxt + n * 96 + seg * 16));
        }
      }
      v16h a = frag_a(ldsH1, H1STR, 16 * w, ks * 32, lane);
#pragma unroll
      for (int t = 0; t < 8; ++t) {
        v16h b = frag_bt(cur, BSTR, 0, 16 * t, lane);
        acc[t] = wmma_f16(a, b, acc[t]);
      }
      if (ks < 7) { async_wait(); __syncthreads(); }
    }
    __syncthreads();
    float* f = (float*)(smem + OFF_A);
#pragma unroll
    for (int t = 0; t < 8; ++t) {
      int col = 16 * t + (lane & 15);
      float bb = kp.nv_b2[col];
#pragma unroll
      for (int j = 0; j < 8; ++j) {
        int row = 16 * w + j + ((lane < 16) ? 0 : 8);
        f[row * FSTR + col] = fmaxf(acc[t][j] + bb, 0.f);
      }
    }
  }
  __syncthreads();
  if (tid < ROWS) {
    const float* f = (const float*)(smem + OFF_A) + tid * FSTR;
    float s = kp.nv_b3[0];
    for (int c = 0; c < H2DIM; ++c) s += f[c] * kp.nv_w3[c];
    novf[tid] = 1.f / (1.f + expf(-s));
  }
  __syncthreads();

  // ---- P4: build padded all_scores matrix [64x32] f16 ----
  if (tid < ROWS) {
    _Float16* sa = ldsSA + tid * SASTR;
    size_t i = (size_t)(r0 + tid) * NM;
    sa[0] = (_Float16)kp.trad[i + 0];
    sa[1] = (_Float16)kp.trad[i + 1];
    sa[2] = (_Float16)kp.trad[i + 2];
    sa[3] = (_Float16)genf[tid];
    sa[4] = (_Float16)persf[tid];
    for (int c = 5; c < 32; ++c) sa[c] = (_Float16)0.f;
  }
  __syncthreads();

  // ---- P5: score-fusion L1 [64x32]x[32x256] + LayerNorm + relu ----
  {
    v8f acc[16];
#pragma unroll
    for (int t = 0; t < 16; ++t) acc[t] = (v8f)0.0f;
    v16h a = frag_a(ldsSA, SASTR, 16 * w, 0, lane);
#pragma unroll
    for (int t = 0; t < 16; ++t) {
      v16h b = frag_bt(kp.sf1T, 32, 0, 16 * t, lane);
      acc[t] = wmma_f16(a, b, acc[t]);
    }
    float* f = (float*)(smem + OFF_A);
#pragma unroll
    for (int t = 0; t < 16; ++t) {
      int col = 16 * t + (lane & 15);
      float bb = kp.sf_b1[col];
#pragma unroll
      for (int j = 0; j < 8; ++j) {
        int row = 16 * w + j + ((lane < 16) ? 0 : 8);
        f[row * FSTR + col] = acc[t][j] + bb;
      }
    }
  }
  __syncthreads();
  if (tid < ROWS) {
    float* f = (float*)(smem + OFF_A) + tid * FSTR;
    float su = 0.f, sq = 0.f;
    for (int c = 0; c < HDIM; ++c) { float x = f[c]; su += x; sq += x * x; }
    float mu = su * (1.f / HDIM);
    float rs = rsqrtf(fmaxf(sq * (1.f / HDIM) - mu * mu, 0.f) + 1e-5f);
    _Float16* h1 = ldsH1 + tid * H1STR;
    for (int c = 0; c < HDIM; ++c) {
      float o = (f[c] - mu) * rs * kp.sf_g1[c] + kp.sf_be1[c];
      h1[c] = (_Float16)fmaxf(o, 0.f);
    }
  }
  __syncthreads();

  // ---- P6: score-fusion L2 [64x256]x[256x128] + LN + relu + sigmoid head ----
  {
    v8f acc[8];
#pragma unroll
    for (int t = 0; t < 8; ++t) acc[t] = (v8f)0.0f;
    for (int i = tid; i < 512; i += TPB) {
      int n = i >> 2, seg = i & 3;
      async_copy16(kp.sf2T + n * HDIM + seg * 8,
                   (_Float16*)(smem + OFF_B0 + n * 96 + seg * 16));
    }
    async_wait();
    __syncthreads();
    for (int ks = 0; ks < 8; ++ks) {
      _Float16* cur = (_Float16*)(smem + OFF_B0 + (ks & 1) * BBYTES);
      if (ks < 7) {
        char* nxt = smem + OFF_B0 + ((ks + 1) & 1) * BBYTES;
        for (int i = tid; i < 512; i += TPB) {
          int n = i >> 2, seg = i & 3;
          async_copy16(kp.sf2T + n * HDIM + (ks + 1) * 32 + seg * 8,
                       (_Float16*)(nxt + n * 96 + seg * 16));
        }
      }
      v16h a = frag_a(ldsH1, H1STR, 16 * w, ks * 32, lane);
#pragma unroll
      for (int t = 0; t < 8; ++t) {
        v16h b = frag_bt(cur, BSTR, 0, 16 * t, lane);
        acc[t] = wmma_f16(a, b, acc[t]);
      }
      if (ks < 7) { async_wait(); __syncthreads(); }
    }
    __syncthreads();
    float* f = (float*)(smem + OFF_A);
#pragma unroll
    for (int t = 0; t < 8; ++t) {
      int col = 16 * t + (lane & 15);
      float bb = kp.sf_b2[col];
#pragma unroll
      for (int j = 0; j < 8; ++j) {
        int row = 16 * w + j + ((lane < 16) ? 0 : 8);
        f[row * FSTR + col] = acc[t][j] + bb;
      }
    }
  }
  __syncthreads();
  if (tid < ROWS) {
    const float* f = (const float*)(smem + OFF_A) + tid * FSTR;
    float su = 0.f, sq = 0.f;
    for (int c = 0; c < H2DIM; ++c) { float x = f[c]; su += x; sq += x * x; }
    float mu = su * (1.f / H2DIM);
    float rs = rsqrtf(fmaxf(sq * (1.f / H2DIM) - mu * mu, 0.f) + 1e-5f);
    float s = 0.f;
    for (int c = 0; c < H2DIM; ++c) {
      float o = fmaxf((f[c] - mu) * rs * kp.sf_g2[c] + kp.sf_be2[c], 0.f);
      s += o * kp.sf_w3[c];
    }
    finf[tid] = 1.f / (1.f + expf(-(s + kp.sf_b3[0])));
  }
  __syncthreads();

  // ---- P7: confidence calibrator [64x32]x[32x128] + head ----
  {
    v8f acc[8];
#pragma unroll
    for (int t = 0; t < 8; ++t) acc[t] = (v8f)0.0f;
    v16h a = frag_a(ldsSA, SASTR, 16 * w, 0, lane);
#pragma unroll
    for (int t = 0; t < 8; ++t) {
      v16h b = frag_bt(kp.cc1T, 32, 0, 16 * t, lane);
      acc[t] = wmma_f16(a, b, acc[t]);
    }
    float* f = (float*)(smem + OFF_A);
#pragma unroll
    for (int t = 0; t < 8; ++t) {
      int col = 16 * t + (lane & 15);
      float bb = kp.cc_b1[col];
#pragma unroll
      for (int j = 0; j < 8; ++j) {
        int row = 16 * w + j + ((lane < 16) ? 0 : 8);
        f[row * FSTR + col] = fmaxf(acc[t][j] + bb, 0.f);
      }
    }
  }
  __syncthreads();
  if (tid < ROWS) {
    const float* f = (const float*)(smem + OFF_A) + tid * FSTR;
    float s = kp.cc_b2[0];
    for (int c = 0; c < H2DIM; ++c) s += f[c] * kp.cc_w2[c];
    conff[tid] = 1.f / (1.f + expf(-s));
  }
  __syncthreads();

  // ---- P8: outputs: rows of [5, N] ----
  if (tid < ROWS) {
    size_t i = (size_t)r0 + tid;
    kp.out[i]                      = finf[tid];
    kp.out[(size_t)NCAND + i]      = conff[tid];
    kp.out[2 * (size_t)NCAND + i]  = novf[tid];
    kp.out[3 * (size_t)NCAND + i]  = genf[tid];
    kp.out[4 * (size_t)NCAND + i]  = persf[tid];
  }
}

// ============================ host launcher ============================
extern "C" void kernel_launch(void* const* d_in, const int* in_sizes, int n_in,
                              void* d_out, int out_size, void* d_ws, size_t ws_size,
                              hipStream_t stream) {
  (void)in_sizes; (void)out_size; (void)ws_size;
  KParams kp{};
  kp.q    = (const float*)d_in[0];
  kp.cand = (const float*)d_in[1];
  kp.user = (const float*)d_in[2];
  kp.ctx  = (const float*)d_in[3];
  kp.trad = (const float*)d_in[4];
  kp.gen  = (const float*)d_in[5];

  // params flattened alphabetically (jax tree order); support both separate
  // pointers (n_in >= 30) and a single concatenated buffer at d_in[6].
  static const long long psz[24] = {128, 1, 640, 128,            // cc_b1,cc_b2,cc_w1,cc_w2
                                    256, 3, 148224, 768,         // ml_b1,ml_b2,ml_w1,ml_w2
                                    256, 128, 1, 262144, 32768, 128,  // nv_b1,b2,b3,w1,w2,w3
                                    256, 128, 1, 256, 128, 256, 128,  // sf_b1,b2,b3,be1,be2,g1,g2
                                    1280, 32768, 128};                // sf_w1,w2,w3
  const float* P[24];
  if (n_in >= 30) {
    for (int i = 0; i < 24; ++i) P[i] = (const float*)d_in[6 + i];
  } else {
    const float* base = (const float*)d_in[6];
    long long off = 0;
    for (int i = 0; i < 24; ++i) { P[i] = base + off; off += psz[i]; }
  }
  kp.cc_b1 = P[0];  kp.cc_b2 = P[1];  kp.cc_w1 = P[2];  kp.cc_w2 = P[3];
  kp.ml_b1 = P[4];  kp.ml_b2 = P[5];  kp.ml_w1 = P[6];  kp.ml_w2 = P[7];
  kp.nv_b1 = P[8];  kp.nv_b2 = P[9];  kp.nv_b3 = P[10];
  kp.nv_w1 = P[11]; kp.nv_w2 = P[12]; kp.nv_w3 = P[13];
  kp.sf_b1 = P[14]; kp.sf_b2 = P[15]; kp.sf_b3 = P[16];
  kp.sf_be1 = P[17]; kp.sf_be2 = P[18]; kp.sf_g1 = P[19]; kp.sf_g2 = P[20];
  kp.sf_w1 = P[21]; kp.sf_w2 = P[22]; kp.sf_w3 = P[23];

  char* ws = (char*)d_ws;
  kp.w1cT = (_Float16*)(ws + 0);        // 262144 B
  kp.w2T  = (_Float16*)(ws + 262144);   //  65536 B
  kp.gT   = (_Float16*)(ws + 327680);   //  32768 B
  kp.sf1T = (_Float16*)(ws + 360448);   //  16384 B
  kp.sf2T = (_Float16*)(ws + 376832);   //  65536 B
  kp.cc1T = (_Float16*)(ws + 442368);   //   8192 B
  kp.qh1  = (float*)(ws + 450560);
  kp.invg = (float*)(ws + 451584);
  kp.tsum = (float*)(ws + 451712);
  kp.out  = (float*)d_out;

  hipLaunchKernelGGL(k_zero,    dim3(1),            dim3(32),  0, stream, kp);
  hipLaunchKernelGGL(k_tsum,    dim3(256),          dim3(256), 0, stream, kp);
  hipLaunchKernelGGL(k_norms,   dim3(NGEN + 1),     dim3(256), 0, stream, kp);
  hipLaunchKernelGGL(k_prep,    dim3(1),            dim3(256), 0, stream, kp);
  hipLaunchKernelGGL(k_convert, dim3(256),          dim3(256), 0, stream, kp);
  hipLaunchKernelGGL(k_main,    dim3(NCAND / ROWS), dim3(TPB), 0, stream, kp);
}
